// Generator_v4_49134425866662
// MI455X (gfx1250) — matrix operand order repaired and checked by reading, not compiled, not run on hardware
//
#include <hip/hip_runtime.h>
#include <hip/hip_bf16.h>
#include <stdint.h>

// ---------------------------------------------------------------------------
// MI455X / gfx1250 implementation.
//  - All convs lowered to WMMA f16 (f32 accum), weights held as A-fragments.
//  - GRU r/z gates fused into one M=16 WMMA; candidate conv in M rows 0-7.
//  - Upsample+conv via 4-phase 2x2 filters; 1x1 shortcut folded into spare
//    M rows; stride-1 tail convs via banded multi-row A (full M utilization
//    even for the 1-channel W2c conv).
//  - Tile staging global->LDS: TDM (tensor_load_to_lds, TENSORcnt) in RB2,
//    async-to-LDS (ASYNCcnt) in RB1.
// ---------------------------------------------------------------------------

typedef __attribute__((ext_vector_type(16))) _Float16 v16h;
typedef __attribute__((ext_vector_type(8)))  float    v8f;
typedef unsigned int u32x4 __attribute__((ext_vector_type(4)));
typedef int          i32x4 __attribute__((ext_vector_type(4)));
typedef int          i32x8 __attribute__((ext_vector_type(8)));

#define USE_ASYNC_LDS 1

#if defined(__HIP_DEVICE_COMPILE__) && defined(__has_builtin)
#if __has_builtin(__builtin_amdgcn_tensor_load_to_lds) && \
    __has_builtin(__builtin_amdgcn_s_wait_tensorcnt)
#define HAVE_TDM 1
#endif
#endif

__device__ __forceinline__ v8f wmma16(v16h a, v16h b, v8f c) {
  return __builtin_amdgcn_wmma_f32_16x16x32_f16(false, a, false, b, (short)0, c,
                                                false, false);
}

// A-fragment: slot-indexed [32 lanes][16 f16] blobs precomputed in ws.
__device__ __forceinline__ v16h load_afrag(const _Float16* af, int slot) {
  return *(const v16h*)(af + ((size_t)slot * 32 + (threadIdx.x & 31)) * 16);
}

// B-fragment gather from an LDS halo buffer through an offset table.
// B 32x16 f16 layout: col = lane & 15; element e of v16h ->
// K = (e>>3)*16 + half*8 + (e&7).
__device__ __forceinline__ v16h gatherB(const _Float16* __restrict__ buf,
                                        const unsigned short* __restrict__ tab,
                                        int kbase, int base) {
  const int half = (threadIdx.x >> 4) & 1;
  v16h r;
#pragma unroll
  for (int e = 0; e < 16; ++e)
    r[e] = buf[base + (int)tab[kbase + ((e >> 3) << 4) + (half << 3) + (e & 7)]];
  return r;
}

__device__ __forceinline__ void async_copy16B(const _Float16* g, _Float16* l) {
#if USE_ASYNC_LDS
  unsigned lds = (unsigned)(uintptr_t)l;  // low 32 bits of generic = LDS offset
  asm volatile("global_load_async_to_lds_b128 %0, %1, off" ::"v"(lds), "v"(g)
               : "memory");
#else
  *(uint4*)l = *(const uint4*)g;
#endif
}
__device__ __forceinline__ void async_wait() {
#if USE_ASYNC_LDS
  asm volatile("s_wait_asynccnt 0" ::: "memory");
#endif
}

// TDM: DMA a contiguous run of `nelem` 2-byte elements global->LDS via a 1-D
// tensor descriptor. Call from ONE wave only (TDM ignores EXEC); follow with a
// workgroup barrier before other waves read the LDS destination.
__device__ __forceinline__ void tdm_load_2B(const void* gsrc, void* ldst,
                                            int nelem) {
#ifdef HAVE_TDM
  const unsigned lds = (unsigned)(uintptr_t)ldst;
  const unsigned long long ga = (unsigned long long)(uintptr_t)gsrc;
  u32x4 g0;
  g0[0] = 1u;                                   // count=1, user descriptor
  g0[1] = lds;                                  // lds_addr (bytes)
  g0[2] = (unsigned)(ga & 0xffffffffu);         // global_addr[31:0]
  g0[3] = (unsigned)((ga >> 32) & 0x01ffffffu)  // global_addr[56:32]
          | (2u << 30);                         // type = 2 ("image")
  i32x8 g1;
  g1[0] = 1 << 16;                       // data_size=1 (2B), mask=0, flags=0
  g1[1] = (nelem & 0xffff) << 16;        // tensor_dim0[15:0]  (bits 63:48)
  g1[2] = ((nelem >> 16) & 0xffff)       // tensor_dim0[31:16]
          | (1 << 16);                   // tensor_dim1 = 1    (bits 111:80)
  g1[3] = (nelem & 0xffff) << 16;        // tile_dim0 = nelem  (bits 127:112)
  g1[4] = 1;                             // tile_dim1 = 1
  g1[5] = nelem;                         // tensor_dim0_stride[31:0]
  g1[6] = 0;
  g1[7] = 0;
  i32x4 gz = {0, 0, 0, 0};
#if __clang_major__ >= 23
  i32x8 gz8 = {0, 0, 0, 0, 0, 0, 0, 0};
  __builtin_amdgcn_tensor_load_to_lds(g0, g1, gz, gz, gz8, 0);
#else
  __builtin_amdgcn_tensor_load_to_lds(g0, g1, gz, gz, 0);
#endif
  __builtin_amdgcn_s_wait_tensorcnt(0);
#else
  (void)gsrc; (void)ldst; (void)nelem;
#endif
}

// phase row-map: which union-window row a 3-tap index hits for phase d.
__device__ __forceinline__ int rmap(int d, int t) {
  return (d == 0) ? (t == 0 ? 0 : 1) : (t < 2 ? 1 : 2);
}

// ---------------------------------------------------------------------------
// Kernel 0: bake all WMMA A-fragments (weights) into ws.
// A 16x32 f16 layout: lanes0-15 M=lane, lanes16-31 M=lane-16;
// element e -> k_local = (e>>3)*16 + half*8 + (e&7).
// Slots: 0-2 GRU gates | 3-5 GRU cand | 6-10 RB1 gemm1 b-phases |
//        11 RB1 shortcut (k-iter 2) | 12-14 RB1 gemm2 multirow W1c |
//        15-17 RB2 gemm1 (b-phases rows0-7 + shortcut rows8-15) |
//        18-19 RB2 gemm2 multirow W2c.
// ---------------------------------------------------------------------------
__global__ __launch_bounds__(32) void afrag_build(
    const float* __restrict__ Wr, const float* __restrict__ Wz,
    const float* __restrict__ Wc, const float* __restrict__ W1b,
    const float* __restrict__ W1a, const float* __restrict__ W1c,
    const float* __restrict__ W2b, const float* __restrict__ W2a,
    const float* __restrict__ W2c, _Float16* __restrict__ af) {
  const int slot = blockIdx.x;
  const int lane = threadIdx.x;
  const int half = lane >> 4;
  const int m = lane & 15;

#pragma unroll
  for (int e = 0; e < 16; ++e) {
    const int kl = ((e >> 3) << 4) + (half << 3) + (e & 7);
    float v = 0.f;
    if (slot < 3) {                       // GRU gates: K = tap*9 + c (81)
      int k = slot * 32 + kl;
      if (k < 81) v = (m < 8) ? Wr[k * 8 + m] : Wz[k * 8 + (m - 8)];
    } else if (slot < 6) {                // GRU candidate
      int k = (slot - 3) * 32 + kl;
      if (k < 81 && m < 8) v = Wc[k * 8 + m];
    } else if (slot < 11) {               // RB1 gemm1 b-part, K=(ry*3+rx)*16+c
      int k = (slot - 6) * 32 + kl;
      if (k < 144) {
        int ry = k / 48, rx = (k / 16) % 3, c = k & 15;
        if (c < 8) {                      // relu(x) channels only
          int o = m & 3, dy = m >> 3, dx = (m >> 2) & 1;
          for (int ky = 0; ky < 3; ++ky)
            for (int kx = 0; kx < 3; ++kx)
              if (rmap(dy, ky) == ry && rmap(dx, kx) == rx)
                v += W1b[((ky * 3 + kx) * 8 + c) * 4 + o];
        }
      }
    } else if (slot == 11) {              // RB1 shortcut: center tap, x-chans
      int k = 64 + kl;
      if (k >= 72 && k < 80) v = W1a[(k - 72) * 4 + (m & 3)];
    } else if (slot < 15) {               // RB1 gemm2 multirow W1c
      int k = (slot - 12) * 32 + kl;
      if (k < 72) {
        int ry = k / 12, rx = (k / 4) % 3, c = k & 3;
        int s = m >> 2, o = m & 3, ky = ry - s;
        if (ky >= 0 && ky < 3) v = W1c[((ky * 3 + rx) * 4 + c) * 4 + o];
      }
    } else if (slot < 18) {               // RB2 gemm1, K=(ry*3+rx)*8+c (96)
      int k = (slot - 15) * 32 + kl;
      int ry = k / 24, rx = (k / 8) % 3, c = k & 7;
      if (m < 8) {                        // b-part: rowshift s, phase
        int s = m >> 2, ph = m & 3, dy = ph >> 1, dx = ph & 1;
        if (c < 4)
          for (int ky = 0; ky < 3; ++ky)
            for (int kx = 0; kx < 3; ++kx)
              if (s + rmap(dy, ky) == ry && rmap(dx, kx) == rx)
                v += W2b[(ky * 3 + kx) * 4 + c];
      } else {                            // a-part: center tap, raw channels
        int s = (m - 8) >> 2;
        if (ry == s + 1 && rx == 1 && c >= 4) v = W2a[c - 4];
      }
    } else {                              // RB2 gemm2 multirow W2c (54)
      int k = (slot - 18) * 32 + kl;
      if (k < 54) {
        int ry = k / 3, rx = k % 3, ky = ry - m;
        if (ky >= 0 && ky < 3) v = W2c[ky * 3 + rx];
      }
    }
    af[((size_t)slot * 32 + lane) * 16 + e] = (_Float16)v;
  }
}

// ---------------------------------------------------------------------------
// Kernel 1: dense + relu, x = relu(z @ Wd + bd) -> f16.  One wave per 16x16
// tile; K=100 padded to 128 (4 k-iters).  Boundary handled with clamped
// addresses + value select (no EXEC-masked loads near WMMA).
// ---------------------------------------------------------------------------
__global__ __launch_bounds__(256) void dense_kernel(
    const float* __restrict__ z, const float* __restrict__ Wd,
    const float* __restrict__ bd, _Float16* __restrict__ xf16) {
  const int wid = (blockIdx.x * 256 + threadIdx.x) >> 5;  // 0..255
  const int lane = threadIdx.x & 31;
  const int half = lane >> 4, idx = lane & 15;
  const int tm = wid >> 4, tn = wid & 15;
  v8f acc = {};
#pragma unroll
  for (int i = 0; i < 4; ++i) {
    v16h a, b;
    const int row = tm * 16 + idx;
    const int col = tn * 16 + idx;
#pragma unroll
    for (int e = 0; e < 16; ++e) {
      int ka = i * 32 + ((e >> 3) << 4) + (half << 3) + (e & 7);
      float za = z[row * 100 + (ka < 100 ? ka : 0)];   // clamped, uncond load
      a[e] = (_Float16)(ka < 100 ? za : 0.f);
      int kb = i * 32 + ((e >> 3) << 4) + (half << 3) + (e & 7);
      float wb = Wd[(kb < 100 ? kb : 0) * 256 + col];
      b[e] = (_Float16)(kb < 100 ? wb : 0.f);
    }
    acc = wmma16(a, b, acc);
  }
#pragma unroll
  for (int j = 0; j < 8; ++j) {
    int row = tm * 16 + j + 8 * half;
    int col = tn * 16 + idx;
    xf16[row * 256 + col] = (_Float16)fmaxf(acc[j] + bd[col], 0.f);
  }
}

// ---------------------------------------------------------------------------
// Kernel 2: ConvGRU, 64 steps. One workgroup (4 waves) per image.
// cat  = [inp(ch0), state(ch1-8)]   -> gates (r|z fused, M=16)
// cat2 = [state*r(ch0-7), inp(ch8)] -> candidate (M rows 0-7)
// ---------------------------------------------------------------------------
#define CATSZ (18 * 18 * 9)  // 2916, +zero-slot

__global__ __launch_bounds__(128) void gru_kernel(
    const _Float16* __restrict__ xf16, const _Float16* __restrict__ af,
    const float* __restrict__ br, const float* __restrict__ bz,
    const float* __restrict__ bc, _Float16* __restrict__ outs) {
  __shared__ _Float16 cat[CATSZ + 4];
  __shared__ _Float16 cat2[CATSZ + 4];
  __shared__ float state[16 * 16 * 8];
  __shared__ unsigned short tab[96];

  const int b = blockIdx.x;
  const int tid = threadIdx.x;
  const int lane = tid & 31, wv = tid >> 5;
  const int half = lane >> 4, n = lane & 15;

  for (int i = tid; i < CATSZ + 4; i += 128) {
    cat[i] = (_Float16)0.f;
    cat2[i] = (_Float16)0.f;
  }
  for (int i = tid; i < 2048; i += 128) state[i] = 0.f;
  if (tid < 96) {
    int k = tid;
    unsigned short off = (unsigned short)CATSZ;  // zero slot for padded K
    if (k < 81) {
      int tap = k / 9, c = k % 9, ky = tap / 3, kx = tap % 3;
      off = (unsigned short)((ky * 18 + kx) * 9 + c);
    }
    tab[k] = off;
  }
  __syncthreads();
  for (int p = tid; p < 256; p += 128) {  // constant input channel
    int y = p >> 4, x = p & 15;
    _Float16 v = xf16[b * 256 + p];
    cat[((y + 1) * 18 + (x + 1)) * 9 + 0] = v;
    cat2[((y + 1) * 18 + (x + 1)) * 9 + 8] = v;
  }

  v16h AG0 = load_afrag(af, 0), AG1 = load_afrag(af, 1), AG2 = load_afrag(af, 2);
  v16h AC0 = load_afrag(af, 3), AC1 = load_afrag(af, 4), AC2 = load_afrag(af, 5);
  float brz[8], bcv[8];
#pragma unroll
  for (int j = 0; j < 8; ++j) {
    brz[j] = half ? bz[j] : br[j];
    bcv[j] = bc[j];
  }
  __syncthreads();

  v8f g[4];
  for (int s = 0; s < 64; ++s) {
    // ---- gates: D rows 0-7 = r (lanes 0-15), rows 8-15 = z (lanes 16-31)
#pragma unroll
    for (int rt = 0; rt < 4; ++rt) {
      const int row = wv * 4 + rt;
      const int base = (row * 18 + n) * 9;
      v8f acc = {};
      acc = wmma16(AG0, gatherB(cat, tab, 0, base), acc);
      acc = wmma16(AG1, gatherB(cat, tab, 32, base), acc);
      acc = wmma16(AG2, gatherB(cat, tab, 64, base), acc);
#pragma unroll
      for (int j = 0; j < 8; ++j)
        acc[j] = 1.f / (1.f + __expf(-(acc[j] + brz[j])));
      g[rt] = acc;
      if (half == 0) {  // cat2 ch0-7 = state * r
#pragma unroll
        for (int j = 0; j < 8; ++j)
          cat2[base + 171 + j] = (_Float16)(state[(row * 16 + n) * 8 + j] * acc[j]);
      }
    }
    __syncthreads();
    // ---- candidate + state update
#pragma unroll
    for (int rt = 0; rt < 4; ++rt) {
      const int row = wv * 4 + rt;
      const int base = (row * 18 + n) * 9;
      v8f acc = {};
      acc = wmma16(AC0, gatherB(cat2, tab, 0, base), acc);
      acc = wmma16(AC1, gatherB(cat2, tab, 32, base), acc);
      acc = wmma16(AC2, gatherB(cat2, tab, 64, base), acc);
      _Float16 hh[8];
#pragma unroll
      for (int j = 0; j < 8; ++j) {
        float zj = __shfl(g[rt][j], (lane & 15) + 16, 32);  // z lives in hi half
        float hc = fmaxf(acc[j] + bcv[j], 0.f);
        float st = state[(row * 16 + n) * 8 + j];
        float h = zj * hc + (1.f - zj) * st;
        hh[j] = (_Float16)h;
        if (half == 0) {
          state[(row * 16 + n) * 8 + j] = h;
          cat[base + 172 + j] = hh[j];  // cat ch 1+j
        }
      }
      if (half == 0) {
        union { _Float16 h[8]; uint4 u; } pk;
#pragma unroll
        for (int j = 0; j < 8; ++j) pk.h[j] = hh[j];
        *(uint4*)&outs[(((size_t)b * 64 + s) * 256 + row * 16 + n) * 8] = pk.u;
      }
    }
    __syncthreads();
  }
}

// ---------------------------------------------------------------------------
// Kernel 3: ResBlockUp2D(4). One workgroup per image (16384).
// GEMM1: 4-phase upsample-conv, M = phase*4+outch; shortcut in separate acc.
// GEMM2: banded multirow W1c, M = rowshift*4+outch.
// Tile staging via async-to-LDS (ASYNCcnt).
// ---------------------------------------------------------------------------
#define B1SZ (34 * 34 * 16)  // 18496
#define B2SZ (34 * 34 * 4)   // 4624

__global__ __launch_bounds__(128) void rb1_kernel(
    const _Float16* __restrict__ outs, const _Float16* __restrict__ af,
    const float* __restrict__ b1a, const float* __restrict__ b1b,
    const float* __restrict__ b1c, _Float16* __restrict__ r1out) {
  __shared__ _Float16 xb[2048];
  __shared__ _Float16 buf1[B1SZ + 8];
  __shared__ _Float16 buf2[B2SZ + 8];
  __shared__ _Float16 abuf[32 * 32 * 4];
  __shared__ unsigned short tab1[160];
  __shared__ unsigned short tab2[96];

  const size_t img = blockIdx.x;
  const int tid = threadIdx.x, lane = tid & 31, wv = tid >> 5;
  const int half = lane >> 4, n = lane & 15;

  const _Float16* src = outs + img * 2048;
  for (int i = tid; i < 256; i += 128) async_copy16B(src + i * 8, xb + i * 8);
  async_wait();

  if (tid < 160) {
    int k = tid;
    tab1[k] = (k < 144)
                  ? (unsigned short)((k / 48) * 34 * 16 + ((k / 16) % 3) * 16 + (k & 15))
                  : (unsigned short)B1SZ;
  }
  if (tid < 96) {
    int k = tid;
    tab2[k] = (k < 72)
                  ? (unsigned short)((k / 12) * 34 * 4 + ((k / 4) % 3) * 4 + (k & 3))
                  : (unsigned short)B2SZ;
  }
  for (int i = tid; i < B2SZ + 8; i += 128) buf2[i] = (_Float16)0.f;
  if (tid == 0) buf1[B1SZ] = (_Float16)0.f;
  __syncthreads();

  // buf1: padded up2 coords, ch0-7 relu(x), ch8-15 raw x
  for (int p = tid; p < 1156; p += 128) {
    int py = p / 34, px = p % 34;
    bool inter = (py >= 1 && py <= 32 && px >= 1 && px <= 32);
    int oy = (py - 1) >> 1, ox = (px - 1) >> 1;
#pragma unroll
    for (int c = 0; c < 8; ++c) {
      float t = inter ? (float)xb[((oy << 4) + ox) * 8 + c] : 0.f;
      buf1[p * 16 + c] = (_Float16)fmaxf(t, 0.f);
      buf1[p * 16 + 8 + c] = (_Float16)t;
    }
  }
  __syncthreads();

  float b1bv[4], b1av[4], b1cv[4];
#pragma unroll
  for (int o = 0; o < 4; ++o) {
    b1bv[o] = b1b[o];
    b1av[o] = b1a[o];
    b1cv[o] = b1c[o];
  }
  v16h A_B[5];
#pragma unroll
  for (int i = 0; i < 5; ++i) A_B[i] = load_afrag(af, 6 + i);
  const v16h A_A = load_afrag(af, 11);
  v16h A_C[3];
#pragma unroll
  for (int i = 0; i < 3; ++i) A_C[i] = load_afrag(af, 12 + i);

  // GEMM1: 16 orig rows, 4 per wave
#pragma unroll
  for (int t = 0; t < 4; ++t) {
    const int y = wv * 4 + t;
    const int base = (y * 34 + n) * 16;
    v8f Db = {}, Da = {};
#pragma unroll
    for (int i = 0; i < 5; ++i)
      Db = wmma16(A_B[i], gatherB(buf1, tab1, i * 32, base), Db);
    Da = wmma16(A_A, gatherB(buf1, tab1, 64, base), Da);
#pragma unroll
    for (int j = 0; j < 8; ++j) {
      int m = j + 8 * half;
      int o = m & 3, ph = m >> 2, dy = ph >> 1, dx = ph & 1;
      int Y = 2 * y + dy, X = 2 * n + dx;
      buf2[((Y + 1) * 34 + (X + 1)) * 4 + o] =
          (_Float16)fmaxf(Db[j] + b1bv[o], 0.f);
      abuf[(Y * 32 + X) * 4 + o] = (_Float16)(Da[j] + b1av[o]);
    }
  }
  __syncthreads();

  // GEMM2: multirow, 16 tiles (8 row-groups x 2 col-tiles), 4 per wave
#pragma unroll
  for (int t = 0; t < 4; ++t) {
    const int tile = wv * 4 + t;
    const int g = tile >> 1, xt = tile & 1;
    const int X = xt * 16 + n;
    const int base = ((4 * g) * 34 + X) * 4;
    v8f D = {};
#pragma unroll
    for (int i = 0; i < 3; ++i)
      D = wmma16(A_C[i], gatherB(buf2, tab2, i * 32, base), D);
#pragma unroll
    for (int j = 0; j < 8; ++j) {
      int m = j + 8 * half, sft = m >> 2, o = m & 3;
      int Y = 4 * g + sft;
      float v = D[j] + b1cv[o] + (float)abuf[(Y * 32 + X) * 4 + o];
      r1out[(img * 1024 + Y * 32 + X) * 4 + o] = (_Float16)v;
    }
  }
}

// ---------------------------------------------------------------------------
// Kernel 4: ResBlockUp2D(1) -> final f32 output. One workgroup per image.
// GEMM1: rowshift x phase packing, b-part rows 0-7 + shortcut rows 8-15, K=96.
// GEMM2: banded multirow W2c (M=16 output rows per WMMA pair).
// Tile staging via TDM (TENSORcnt) when available, else async-to-LDS.
// ---------------------------------------------------------------------------
#define BIN2 (34 * 34 * 8)  // 9248
#define BB2 (66 * 66)       // 4356

__global__ __launch_bounds__(128) void rb2_kernel(
    const _Float16* __restrict__ r1out, const _Float16* __restrict__ af,
    const float* __restrict__ b2a, const float* __restrict__ b2b,
    const float* __restrict__ b2c, float* __restrict__ out) {
  __shared__ _Float16 yb[4096];
  __shared__ _Float16 bin[BIN2 + 8];
  __shared__ _Float16 bb1[BB2 + 8];
  __shared__ _Float16 abuf[4096];
  __shared__ unsigned short tabA[96];
  __shared__ unsigned short tabB[64];

  const size_t img = blockIdx.x;
  const int tid = threadIdx.x, lane = tid & 31, wv = tid >> 5;
  const int half = lane >> 4, n = lane & 15;

  const _Float16* src = r1out + img * 4096;
#ifdef HAVE_TDM
  if (wv == 0) tdm_load_2B(src, yb, 4096);  // one wave issues; barrier below
#else
  for (int i = tid; i < 512; i += 128) async_copy16B(src + i * 8, yb + i * 8);
  async_wait();
#endif

  if (tid < 96) {
    int k = tid;  // all K valid: (ry,rx,c) -> padded offset
    tabA[k] = (unsigned short)((k / 24) * 34 * 8 + ((k / 8) % 3) * 8 + (k & 7));
  }
  if (tid < 64) {
    int k = tid;
    tabB[k] = (k < 54) ? (unsigned short)((k / 3) * 66 + k % 3)
                       : (unsigned short)BB2;
  }
  for (int i = tid; i < BB2 + 8; i += 128) bb1[i] = (_Float16)0.f;
  __syncthreads();

  // bin: padded orig coords, ch0-3 relu(y), ch4-7 raw y
  for (int p = tid; p < 1156; p += 128) {
    int py = p / 34, px = p % 34;
    bool inter = (py >= 1 && py <= 32 && px >= 1 && px <= 32);
    int oy = py - 1, ox = px - 1;
#pragma unroll
    for (int c = 0; c < 4; ++c) {
      float t = inter ? (float)yb[((oy << 5) + ox) * 4 + c] : 0.f;
      bin[p * 8 + c] = (_Float16)fmaxf(t, 0.f);
      bin[p * 8 + 4 + c] = (_Float16)t;
    }
  }
  __syncthreads();

  const float vb2b = b2b[0], vb2a = b2a[0], vb2c = b2c[0];
  v16h A1[3];
#pragma unroll
  for (int i = 0; i < 3; ++i) A1[i] = load_afrag(af, 15 + i);
  v16h A2[2];
#pragma unroll
  for (int i = 0; i < 2; ++i) A2[i] = load_afrag(af, 18 + i);

  // GEMM1: 32 tiles (16 row-pairs x 2 col-tiles), 8 per wave
#pragma unroll
  for (int t = 0; t < 8; ++t) {
    const int tile = wv * 8 + t;
    const int yg = tile >> 1, xt = tile & 1;
    const int y = 2 * yg;
    const int x = xt * 16 + n;
    const int base = (y * 34 + x) * 8;
    v8f D = {};
#pragma unroll
    for (int i = 0; i < 3; ++i)
      D = wmma16(A1[i], gatherB(bin, tabA, i * 32, base), D);
#pragma unroll
    for (int j = 0; j < 8; ++j) {
      int m = j + 8 * half;
      int mm = m & 7, sft = mm >> 2, ph = mm & 3, dy = ph >> 1, dx = ph & 1;
      int Y = 2 * (y + sft) + dy, X = 2 * x + dx;
      if (m < 8)  // main path pre-activation
        bb1[(Y + 1) * 66 + (X + 1)] = (_Float16)fmaxf(D[j] + vb2b, 0.f);
      else        // shortcut
        abuf[Y * 64 + X] = (_Float16)(D[j] + vb2a);
    }
  }
  __syncthreads();

  // GEMM2: 16 tiles (4 row-groups x 4 col-tiles), 4 per wave, 2 k-iters
#pragma unroll
  for (int t = 0; t < 4; ++t) {
    const int tile = wv * 4 + t;
    const int g = tile >> 2, xt = tile & 3;
    const int X = xt * 16 + n;
    const int base = (16 * g) * 66 + X;
    v8f D = {};
#pragma unroll
    for (int i = 0; i < 2; ++i)
      D = wmma16(A2[i], gatherB(bb1, tabB, i * 32, base), D);
#pragma unroll
    for (int j = 0; j < 8; ++j) {
      int sft = j + 8 * half;
      int Y = 16 * g + sft;
      out[img * 4096 + Y * 64 + X] = D[j] + vb2c + (float)abuf[Y * 64 + X];
    }
  }
}

// ---------------------------------------------------------------------------
extern "C" void kernel_launch(void* const* d_in, const int* in_sizes, int n_in,
                              void* d_out, int out_size, void* d_ws,
                              size_t ws_size, hipStream_t stream) {
  (void)in_sizes; (void)n_in; (void)out_size; (void)ws_size;
  const float* z   = (const float*)d_in[0];
  const float* Wd  = (const float*)d_in[1];
  const float* bd  = (const float*)d_in[2];
  const float* Wr  = (const float*)d_in[3];
  const float* br  = (const float*)d_in[4];
  const float* Wz  = (const float*)d_in[5];
  const float* bz  = (const float*)d_in[6];
  const float* Wc  = (const float*)d_in[7];
  const float* bc  = (const float*)d_in[8];
  const float* W1a = (const float*)d_in[9];
  const float* b1a = (const float*)d_in[10];
  const float* W1b = (const float*)d_in[11];
  const float* b1b = (const float*)d_in[12];
  const float* W1c = (const float*)d_in[13];
  const float* b1c = (const float*)d_in[14];
  const float* W2a = (const float*)d_in[15];
  const float* b2a = (const float*)d_in[16];
  const float* W2b = (const float*)d_in[17];
  const float* b2b = (const float*)d_in[18];
  const float* W2c = (const float*)d_in[19];
  const float* b2c = (const float*)d_in[20];

  char* ws = (char*)d_ws;
  _Float16* xf16 = (_Float16*)(ws);                          // 128 KB
  _Float16* af   = (_Float16*)(ws + ((size_t)256 << 10));    // 20 KB A-frags
  _Float16* outs = (_Float16*)(ws + ((size_t)1 << 20));      // 64 MB GRU outs
  _Float16* r1o  = (_Float16*)(ws + ((size_t)1 << 20) + ((size_t)1 << 26));

  afrag_build<<<20, 32, 0, stream>>>(Wr, Wz, Wc, W1b, W1a, W1c, W2b, W2a, W2c, af);
  dense_kernel<<<32, 256, 0, stream>>>(z, Wd, bd, xf16);
  gru_kernel<<<256, 128, 0, stream>>>(xf16, af, br, bz, bc, outs);
  rb1_kernel<<<16384, 128, 0, stream>>>(outs, af, b1a, b1b, b1c, r1o);
  rb2_kernel<<<16384, 128, 0, stream>>>(r1o, af, b2a, b2b, b2c, (float*)d_out);
}
